// AxialAttentionBlock_42975442764615
// MI455X (gfx1250) — compile-verified, matrix operand order checked by reading, not run on hardware
//
#include <hip/hip_runtime.h>

#define C_DIM 256
#define S_DIM 65536   // 16*64*64
#define T_DIM 131072  // 2 * S_DIM
#define NHEAD 4
#define DK 64

typedef __attribute__((ext_vector_type(16))) __bf16 v16bf;
typedef __attribute__((ext_vector_type(8)))  float  v8f;
typedef __attribute__((ext_vector_type(4)))  unsigned int v4u;
typedef __attribute__((ext_vector_type(8)))  int v8i;
typedef __attribute__((ext_vector_type(4)))  int v4i;

__device__ __forceinline__ v8f vzero8() {
  v8f z;
#pragma unroll
  for (int i = 0; i < 8; ++i) z[i] = 0.0f;
  return z;
}

// Build a 16-element bf16 fragment for WMMA A (row = M) or B (row = N, holding
// B^T row) from a contiguous 16-bit row in LDS, following the CDNA5
// 16-bit 16x32 layout: lanes 0-15 K = {0..7,16..23}, lanes 16-31 K = {8..15,24..31}.
__device__ __forceinline__ v16bf ldfrag(const __bf16* row, int k0) {
  const int half = (threadIdx.x >> 4) & 1;
  union { unsigned u[8]; v16bf v; } r;
#pragma unroll
  for (int i = 0; i < 8; ++i) {
    const int k = k0 + ((i >> 2) << 4) + (half << 3) + ((i & 3) << 1);
    r.u[i] = *reinterpret_cast<const unsigned*>(row + k);
  }
  return r.v;
}

__device__ __forceinline__ v8f wmma_bf16(v16bf a, v16bf b, v8f c) {
  return __builtin_amdgcn_wmma_f32_16x16x32_bf16(false, a, false, b, (short)0, c,
                                                 false, false);
}

// Tensor Data Mover: async 2D tile load (bf16 elements) global -> LDS.
// Descriptor bitfields per CDNA5 ISA D# groups 0/1 (2D tile: groups 2/3 zero).
// clang-23 toolchain: 6-arg builtin (g0 v4u, g1 v8i, g2 v4i, g3 v4i, g4 v8i, cpol).
__device__ __forceinline__ void tdm_load_2d(const void* gsrc, unsigned lds_off,
                                            int tile_d0, int tile_d1,
                                            int tensor_d0, int tensor_d1,
                                            int stride0_elems) {
  unsigned long long ga = (unsigned long long)(size_t)gsrc;
  v4u g0;
  g0.x = 1u;                                              // count=1 (valid), user mode
  g0.y = lds_off;                                         // lds_addr (bytes)
  g0.z = (unsigned)ga;                                    // global_addr[31:0]
  g0.w = (unsigned)((ga >> 32) & 0x01ffffffu) | (2u << 30); // addr[56:32] | type=2
  v8i g1;
  g1[0] = (int)(1u << 16);                                // data_size=1 -> 2 bytes
  g1[1] = (int)(((unsigned)tensor_d0 & 0xffffu) << 16);   // tensor_dim0[15:0]
  g1[2] = (int)((((unsigned)tensor_d0 >> 16) & 0xffffu) |
                (((unsigned)tensor_d1 & 0xffffu) << 16)); // td0[31:16] | td1[15:0]
  g1[3] = (int)((((unsigned)tensor_d1 >> 16) & 0xffffu) |
                ((unsigned)tile_d0 << 16));               // td1[31:16] | tile_dim0
  g1[4] = (int)(unsigned)tile_d1;                         // tile_dim1 (tile_dim2=0)
  g1[5] = stride0_elems;                                  // tensor_dim0_stride[31:0]
  g1[6] = 0;
  g1[7] = 0;
  v4i gz4 = {0, 0, 0, 0};
  v8i gz8;
#pragma unroll
  for (int i = 0; i < 8; ++i) gz8[i] = 0;
  __builtin_amdgcn_tensor_load_to_lds(g0, g1, gz4, gz4, gz8, 0);
}

// ---------------------------------------------------------------------------
// f32 -> bf16 elementwise conversion (weights)
// ---------------------------------------------------------------------------
__global__ void cvt_bf16(const float* __restrict__ src, __bf16* __restrict__ dst,
                         int n) {
  int i = blockIdx.x * 256 + threadIdx.x;
  if (i < n) dst[i] = (__bf16)src[i];
}

// ---------------------------------------------------------------------------
// Q/K/V projection: C^T[ch][t] = sum_c x[b,c,s(t)] * W[ch][c]
// x channel-major f32, W bf16 [256][256] row-major, output channel-major bf16.
// Block: 256 threads = 8 waves; tile 64 tokens x 256 channels; K step 32.
// B tile staged by the Tensor Data Mover (async, TENSORcnt).
// ---------------------------------------------------------------------------
__global__ __launch_bounds__(256) void proj_gemm(const float* __restrict__ x,
                                                 const __bf16* __restrict__ W,
                                                 __bf16* __restrict__ outc) {
  __shared__ __align__(16) __bf16 As[64 * 32];
  __shared__ __align__(16) __bf16 Bs[256 * 32];
  const int tid = threadIdx.x;
  const int lane = tid & 31;
  const int l15 = lane & 15;
  const int w = tid >> 5;
  const int wm = w & 3;        // M sub-tile (16 tokens)
  const int wn = w >> 2;       // N half (128 channels)
  const long t0 = (long)blockIdx.x * 64;
  const int b = (int)(t0 >> 16);
  const int s0 = (int)(t0 & 65535);

  v8f acc[8];
#pragma unroll
  for (int nf = 0; nf < 8; ++nf) acc[nf] = vzero8();

  for (int k0 = 0; k0 < C_DIM; k0 += 32) {
    // B tile via TDM: 256 rows x 32 (c), issued once (wave 0).
    if (tid < 32) {
      tdm_load_2d(W + k0, (unsigned)(size_t)(const void*)Bs, 32, 256, 256, 256,
                  256);
    }
    // A tile: 64 tokens x 32 channels, f32 -> bf16, coalesced reads from x.
#pragma unroll
    for (int it = 0; it < 8; ++it) {
      int e = it * 256 + tid;            // 0..2047
      int kk = e >> 6;                   // 0..31
      int tt = e & 63;                   // 0..63
      As[tt * 32 + kk] =
          (__bf16)x[(size_t)b * C_DIM * S_DIM + (size_t)(k0 + kk) * S_DIM + s0 + tt];
    }
    if (tid < 32) __builtin_amdgcn_s_wait_tensorcnt(0);
    __syncthreads();

    v16bf a = ldfrag(&As[(wm * 16 + l15) * 32], 0);
#pragma unroll
    for (int nf = 0; nf < 8; ++nf) {
      v16bf bb = ldfrag(&Bs[(wn * 128 + nf * 16 + l15) * 32], 0);
      acc[nf] = wmma_bf16(a, bb, acc[nf]);
    }
    __syncthreads();
  }

  // Store: per lane 8 consecutive tokens of one channel -> one b128 store.
  const int trow = (int)t0 + wm * 16 + ((lane < 16) ? 0 : 8);
#pragma unroll
  for (int nf = 0; nf < 8; ++nf) {
    int ch = wn * 128 + nf * 16 + l15;
    union { __bf16 h[8]; uint4 q; } pk;
#pragma unroll
    for (int r = 0; r < 8; ++r) pk.h[r] = (__bf16)acc[nf][r];
    *reinterpret_cast<uint4*>(outc + (size_t)ch * T_DIM + trow) = pk.q;
  }
}

// ---------------------------------------------------------------------------
// Axial attention: one wave per (batch, head, sequence).
// Q/K/V channel-major bf16; O written channel-major bf16.
// ---------------------------------------------------------------------------
__global__ __launch_bounds__(32) void axial_attn(
    const __bf16* __restrict__ Qc, const __bf16* __restrict__ Kc,
    const __bf16* __restrict__ Vc, __bf16* __restrict__ Oc, int L, int stride,
    int nbases, int divv, int mulHi, int mulLo) {
  __shared__ __align__(16) __bf16 Qs[64 * 64];
  __shared__ __align__(16) __bf16 Ks[64 * 64];
  __shared__ __align__(16) __bf16 VsT[64 * 64];   // [dk][pos]
  __shared__ __align__(16) __bf16 Pb[16 * 64];
  __shared__ __align__(16) float Sb[16 * 64];

  const int lane = threadIdx.x;
  const int l15 = lane & 15;
  const int half_hi = (lane >= 16) ? 8 : 0;
  int bid = blockIdx.x;
  const int baseIdx = bid % nbases;
  const int head = (bid / nbases) & 3;
  const int b = bid / (nbases * 4);
  const int s_base = (baseIdx / divv) * mulHi + (baseIdx % divv) * mulLo;
  const size_t tokBase = (size_t)b * S_DIM + s_base;

  // Stage Q, K (row = pos, col = dk) and V^T (row = dk, col = pos).
  const int nelem = L * DK;
  for (int e = lane; e < nelem; e += 32) {
    int i = e % L;
    int k = e / L;
    size_t g = (size_t)(head * DK + k) * T_DIM + tokBase + (size_t)i * stride;
    Qs[i * 64 + k] = Qc[g];
    Ks[i * 64 + k] = Kc[g];
    VsT[k * 64 + i] = Vc[g];
  }
  if (L == 16) {  // zero-pad K-dim of the PV GEMM to 32
    for (int e = lane; e < 64 * 16; e += 32)
      VsT[(e / 16) * 64 + 16 + (e % 16)] = (__bf16)0.0f;
    for (int e = lane; e < 16 * 16; e += 32)
      Pb[(e / 16) * 64 + 16 + (e % 16)] = (__bf16)0.0f;
  }
  __syncthreads();

  const float scale = 0.125f;  // 1/sqrt(64)
  const int mtiles = L >> 4;
  const int pvk = (L == 16) ? 32 : 64;

  for (int mt = 0; mt < mtiles; ++mt) {
    // S = Q K^T for this 16-row block.
    for (int nt = 0; nt < mtiles; ++nt) {
      v8f acc = vzero8();
      v16bf a0 = ldfrag(&Qs[(mt * 16 + l15) * 64], 0);
      v16bf b0 = ldfrag(&Ks[(nt * 16 + l15) * 64], 0);
      acc = wmma_bf16(a0, b0, acc);
      a0 = ldfrag(&Qs[(mt * 16 + l15) * 64], 32);
      b0 = ldfrag(&Ks[(nt * 16 + l15) * 64], 32);
      acc = wmma_bf16(a0, b0, acc);
#pragma unroll
      for (int r = 0; r < 8; ++r)
        Sb[(r + half_hi) * 64 + nt * 16 + l15] = acc[r] * scale;
    }
    __syncthreads();

    // Row softmax in f32 (one lane per query row).
    if (lane < 16) {
      float mx = -3.4e38f;
      for (int j = 0; j < L; ++j) mx = fmaxf(mx, Sb[lane * 64 + j]);
      float sum = 0.0f;
      for (int j = 0; j < L; ++j) {
        float e = __expf(Sb[lane * 64 + j] - mx);
        sum += e;
        Sb[lane * 64 + j] = e;
      }
      float inv = 1.0f / sum;
      for (int j = 0; j < L; ++j)
        Pb[lane * 64 + j] = (__bf16)(Sb[lane * 64 + j] * inv);
    }
    __syncthreads();

    // O = P V for this row block (dk = 64 -> 4 N tiles).
    for (int nt = 0; nt < 4; ++nt) {
      v8f acc = vzero8();
      for (int kc = 0; kc < pvk; kc += 32) {
        v16bf a0 = ldfrag(&Pb[l15 * 64], kc);
        v16bf b0 = ldfrag(&VsT[(nt * 16 + l15) * 64], kc);
        acc = wmma_bf16(a0, b0, acc);
      }
      int ch = head * DK + nt * 16 + l15;
#pragma unroll
      for (int r = 0; r < 8; ++r) {
        int i = mt * 16 + r + half_hi;
        Oc[(size_t)ch * T_DIM + tokBase + (size_t)i * stride] = (__bf16)acc[r];
      }
    }
    __syncthreads();
  }
}

// ---------------------------------------------------------------------------
// Output projection: out[b,ch,s] (+)= sum_c o[c][t] * Wo[ch][c] + bo[ch]
// ---------------------------------------------------------------------------
__global__ __launch_bounds__(256) void outproj_gemm(
    const __bf16* __restrict__ oc, const __bf16* __restrict__ W,
    const float* __restrict__ bias, float* __restrict__ out, int accFlag) {
  __shared__ __align__(16) __bf16 As[64 * 32];
  __shared__ __align__(16) __bf16 Bs[256 * 32];
  const int tid = threadIdx.x;
  const int lane = tid & 31;
  const int l15 = lane & 15;
  const int w = tid >> 5;
  const int wm = w & 3;
  const int wn = w >> 2;
  const long t0 = (long)blockIdx.x * 64;
  const int b = (int)(t0 >> 16);
  const int s0 = (int)(t0 & 65535);

  v8f acc[8];
#pragma unroll
  for (int nf = 0; nf < 8; ++nf) acc[nf] = vzero8();

  for (int k0 = 0; k0 < C_DIM; k0 += 32) {
    if (tid < 32) {
      tdm_load_2d(W + k0, (unsigned)(size_t)(const void*)Bs, 32, 256, 256, 256,
                  256);
    }
#pragma unroll
    for (int it = 0; it < 8; ++it) {
      int e = it * 256 + tid;
      int kk = e >> 6;
      int tt = e & 63;
      As[tt * 32 + kk] = oc[(size_t)(k0 + kk) * T_DIM + t0 + tt];
    }
    if (tid < 32) __builtin_amdgcn_s_wait_tensorcnt(0);
    __syncthreads();

    v16bf a = ldfrag(&As[(wm * 16 + l15) * 32], 0);
#pragma unroll
    for (int nf = 0; nf < 8; ++nf) {
      v16bf bb = ldfrag(&Bs[(wn * 128 + nf * 16 + l15) * 32], 0);
      acc[nf] = wmma_bf16(a, bb, acc[nf]);
    }
    __syncthreads();
  }

  const int srow = s0 + wm * 16 + ((lane < 16) ? 0 : 8);
#pragma unroll
  for (int nf = 0; nf < 8; ++nf) {
    int ch = wn * 128 + nf * 16 + l15;
    float bch = bias[ch];
    size_t base = (size_t)b * C_DIM * S_DIM + (size_t)ch * S_DIM + srow;
#pragma unroll
    for (int r = 0; r < 8; ++r) {
      float v = acc[nf][r] + bch;
      if (accFlag) v += out[base + r];
      out[base + r] = v;
    }
  }
}

// ---------------------------------------------------------------------------
extern "C" void kernel_launch(void* const* d_in, const int* in_sizes, int n_in,
                              void* d_out, int out_size, void* d_ws,
                              size_t ws_size, hipStream_t stream) {
  const float* x = (const float*)d_in[0];
  const float* Wq = (const float*)d_in[1];
  const float* Wk = (const float*)d_in[2];
  const float* Wv = (const float*)d_in[3];
  const float* Wo = (const float*)d_in[4];
  const float* bo = (const float*)d_in[5];
  float* out = (float*)d_out;

  char* ws = (char*)d_ws;
  const size_t wElems = 3 * 256 * 256;           // per weight set
  __bf16* WbQ = (__bf16*)ws;
  __bf16* WbK = WbQ + wElems;
  __bf16* WbV = WbK + wElems;
  __bf16* WbO = WbV + wElems;
  size_t off = 4 * wElems * sizeof(__bf16);      // 1572864 bytes, 16B aligned
  const size_t qkvBytes = (size_t)T_DIM * C_DIM * sizeof(__bf16);
  __bf16* Qc = (__bf16*)(ws + off);
  __bf16* Kc = (__bf16*)(ws + off + qkvBytes);
  __bf16* Vc = (__bf16*)(ws + off + 2 * qkvBytes);
  __bf16* Oc = (__bf16*)(ws + off + 3 * qkvBytes);

  // Convert weights to bf16.
  cvt_bf16<<<768, 256, 0, stream>>>(Wq, WbQ, (int)wElems);
  cvt_bf16<<<768, 256, 0, stream>>>(Wk, WbK, (int)wElems);
  cvt_bf16<<<768, 256, 0, stream>>>(Wv, WbV, (int)wElems);
  cvt_bf16<<<768, 256, 0, stream>>>(Wo, WbO, (int)wElems);

  // Per-axis attention config: {L, stride, nbases, div, mulHi, mulLo}
  const int cfg[3][6] = {
      {16, 4096, 4096, 64, 64, 1},    // axis 0: along d1
      {64, 64, 1024, 64, 4096, 1},    // axis 1: along d2
      {64, 1, 1024, 64, 4096, 64},    // axis 2: along d3
  };

  const int gemmGrid = T_DIM / 64;  // 2048
  for (int d = 0; d < 3; ++d) {
    proj_gemm<<<gemmGrid, 256, 0, stream>>>(x, WbQ + (size_t)d * 65536, Qc);
    proj_gemm<<<gemmGrid, 256, 0, stream>>>(x, WbK + (size_t)d * 65536, Kc);
    proj_gemm<<<gemmGrid, 256, 0, stream>>>(x, WbV + (size_t)d * 65536, Vc);

    int nblk = 2 * NHEAD * cfg[d][2];
    axial_attn<<<nblk, 32, 0, stream>>>(Qc, Kc, Vc, Oc, cfg[d][0], cfg[d][1],
                                        cfg[d][2], cfg[d][3], cfg[d][4],
                                        cfg[d][5]);

    outproj_gemm<<<gemmGrid, 256, 0, stream>>>(Oc, WbO + (size_t)d * 65536,
                                               bo + (size_t)d * 256, out, d);
  }
}
